// FeatureCorrelationFlow_12343736009129
// MI455X (gfx1250) — matrix operand order repaired and checked by reading, not compiled, not run on hardware
//
#include <hip/hip_runtime.h>
#include <hip/hip_bf16.h>

// ---------------------------------------------------------------------------
// FeatureCorrelationFlow on MI455X (gfx1250, wave32, WMMA)
//
//   1) chw_f32_to_hwc_bf16 : F1/F2 f32 [B,C,H,W] -> bf16 [B,H,W,C] (LDS transpose)
//   2) pack_weights        : w1/w2/w3 f32 -> bf16 A-fragment layout (zero padded)
//   3) corr_kernel         : 81-shift correlation via 16x16 Gram WMMA tiles,
//                            output bf16 [B,H,W,96] (pad channels zeroed)
//   4) conv3x3_wmma        : implicit-GEMM 3x3 convs, pixel-major bf16 I/O
//
// Pixel-major bf16 makes every WMMA operand a contiguous 16/32-byte load;
// depth-1 software pipelining overlaps fragment loads with WMMA issue.
// ---------------------------------------------------------------------------

typedef __attribute__((ext_vector_type(16))) __bf16 v16bf;
typedef __attribute__((ext_vector_type(8)))  __bf16 v8bf;
typedef __attribute__((ext_vector_type(8)))  float  v8f;

#define BATCH 8
#define CHAN  256
#define HH    128
#define WW    128
#define NSH   81    // (2*4+1)^2 shifts
#define CORRP 96    // corr channels padded to 96 for 3 chunks of 32

// f32 -> bf16 bits (round-to-nearest-even); only in bandwidth-bound passes.
static __device__ __forceinline__ unsigned short f32_bf16_bits(float f) {
    unsigned u = __builtin_bit_cast(unsigned, f);
    u += 0x7FFFu + ((u >> 16) & 1u);
    return (unsigned short)(u >> 16);
}

// A-matrix 16x32 layout matching B: lane half h, element e ->
//   K = 16h+e  == one contiguous 16-channel run.
static __device__ __forceinline__ int kmapA(int half, int e) {
    return 16 * half + e;
}
// 16-bit B-matrix 32x16 layout used here: K = 16*half + e == one contiguous run.

struct bfpair { v8bf lo, hi; };
static __device__ __forceinline__ v16bf cat16(v8bf lo, v8bf hi) {
    bfpair p{lo, hi};
    return __builtin_bit_cast(v16bf, p);
}

// ---------------------------------------------------------------------------
// Transpose + downconvert: src f32 [B,C,H,W] -> dst bf16 [B,H,W,C].
// ---------------------------------------------------------------------------
__global__ __launch_bounds__(256)
void chw_f32_to_hwc_bf16(const float* __restrict__ src,
                         unsigned short* __restrict__ dst) {
    __shared__ unsigned short tile[32][72];   // pad to dodge bank conflicts
    int bid = blockIdx.x;                     // B * H * (W/64) * (C/32)
    int cg = bid & 7;
    int wg = (bid >> 3) & 1;
    int r  = (bid >> 4) & 127;
    int b  = bid >> 11;
    int c0 = cg * 32, p0 = wg * 64;
    int t = threadIdx.x;

    int px = t & 63, chq = t >> 6;            // load: coalesced along W
#pragma unroll
    for (int j = 0; j < 8; ++j) {
        int ch = chq * 8 + j;
        float v = src[((size_t)(b * CHAN + c0 + ch) * HH + r) * WW + p0 + px];
        tile[ch][px] = f32_bf16_bits(v);
    }
    __syncthreads();

    int opx = t >> 2, cq = t & 3;             // store: contiguous 16B per thread
    union { unsigned short u[8]; uint4 v; } pk;
#pragma unroll
    for (int j = 0; j < 8; ++j) pk.u[j] = tile[cq * 8 + j][opx];
    *(uint4*)(dst + ((size_t)((b * HH + r) * WW + p0 + opx)) * CHAN + c0 + cq * 8)
        = pk.v;
}

// ---------------------------------------------------------------------------
// Weight packing: w[Cout][Cin][3][3] f32 -> bf16 A fragments
//   packed[(((tap*nchunk + chunk)*nm + mt)*32 + lane)*16 + e]
// ---------------------------------------------------------------------------
__global__ void pack_weights(const float* __restrict__ w,
                             unsigned short* __restrict__ packed,
                             int Cin, int Cout, int nchunk, int nm, int total) {
    int idx = blockIdx.x * 256 + threadIdx.x;
    if (idx >= total) return;
    int e    = idx & 15;
    int t    = idx >> 4;
    int lane = t & 31;  t >>= 5;
    int mt   = t % nm;  t /= nm;
    int chunk = t % nchunk;
    int tap   = t / nchunk;
    int half = lane >> 4;
    int co = mt * 16 + (lane & 15);
    int ci = chunk * 32 + kmapA(half, e);
    float val = 0.0f;
    if (co < Cout && ci < Cin)
        val = w[((size_t)co * Cin + ci) * 9 + tap];
    packed[idx] = f32_bf16_bits(val);
}

// ---------------------------------------------------------------------------
// Correlation: one wave per (b, row, 16-pixel tile).
// G = F1tile(16px x 256ch) * F2row(256ch x 32px), F2 cols p0-8..p0+23 wrapped.
// corr[i, dx] = G[i, i-dx+12] / 16. Output bf16 [B,H,W,96], pads zeroed.
// Depth-1 prefetch of the next chunk's B fragments while WMMAs execute.
// ---------------------------------------------------------------------------
__global__ __launch_bounds__(32)
void corr_kernel(const unsigned short* __restrict__ F1,
                 const unsigned short* __restrict__ F2,
                 unsigned short* __restrict__ corr) {
    __shared__ float G[16 * 32];

    int tile = blockIdx.x;             // B * 128 rows * 8 tiles
    int p0 = (tile & 7) * 16;
    int r  = (tile >> 3) & 127;
    int b  = tile >> 10;
    int lane = threadIdx.x;
    int nm = lane & 15, half = lane >> 4;

    // A fragments: F1 pixels (M=nm) x channels; two 16B loads per chunk.
    v16bf a[8];
    const unsigned short* f1pix =
        F1 + ((size_t)((b * HH + r) * WW) + p0 + nm) * CHAN;
#pragma unroll
    for (int chunk = 0; chunk < 8; ++chunk) {
        v8bf lo = *(const v8bf*)(f1pix + chunk * 32 + 16 * half);
        v8bf hi = *(const v8bf*)(f1pix + chunk * 32 + 16 * half + 8);
        a[chunk] = cat16(lo, hi);
    }

    int colL = (p0 - 8 + nm) & 127;
    int colR = (p0 + 8 + nm) & 127;
    size_t rowpix = (size_t)((b * HH + r) * WW);

    for (int dyi = 0; dyi < 9; ++dyi) {
        int r2 = (r - (dyi - 4)) & 127;            // roll: F2[(h-dy) mod H]
        const unsigned short* f2L =
            F2 + ((size_t)((b * HH + r2) * WW) + colL) * CHAN + 16 * half;
        const unsigned short* f2R =
            F2 + ((size_t)((b * HH + r2) * WW) + colR) * CHAN + 16 * half;
        v8f accL = {}, accR = {};

        v16bf bl = *(const v16bf*)(f2L);
        v16bf br = *(const v16bf*)(f2R);
#pragma unroll
        for (int chunk = 0; chunk < 8; ++chunk) {
            v16bf nbl = bl, nbr = br;
            if (chunk < 7) {                       // prefetch next chunk
                nbl = *(const v16bf*)(f2L + (chunk + 1) * 32);
                nbr = *(const v16bf*)(f2R + (chunk + 1) * 32);
            }
            accL = __builtin_amdgcn_wmma_f32_16x16x32_bf16(
                false, a[chunk], false, bl, (short)0, accL, false, false);
            accR = __builtin_amdgcn_wmma_f32_16x16x32_bf16(
                false, a[chunk], false, br, (short)0, accR, false, false);
            bl = nbl; br = nbr;
        }
        // Spill Gram: rows = F1 pixel, cols = F2 offset 0..31 <-> p0-8..p0+23
#pragma unroll
        for (int v = 0; v < 8; ++v) {
            int m = v + 8 * half;
            G[m * 32 + nm]      = accL[v];
            G[m * 32 + 16 + nm] = accR[v];
        }
        __syncthreads();
        for (int idx = lane; idx < 144; idx += 32) {
            int i   = idx & 15;
            int dxi = idx >> 4;                    // dx = dxi - 4
            float val = G[i * 32 + (i - dxi + 12)] * 0.0625f;
            corr[(rowpix + p0 + i) * CORRP + dyi * 9 + dxi] = f32_bf16_bits(val);
        }
        __syncthreads();
    }
    // Zero the padded channels 81..95 once.
    for (int idx = lane; idx < 16 * (CORRP - NSH); idx += 32) {
        int i = idx / (CORRP - NSH);
        int k = NSH + idx % (CORRP - NSH);
        corr[(rowpix + p0 + i) * CORRP + k] = 0;
    }
}

// ---------------------------------------------------------------------------
// 3x3 conv, implicit GEMM, pixel-major bf16 I/O. One wave per 16-pixel tile.
// Flattened (kx, chunk) steps per valid row with depth-1 A/B prefetch.
// ---------------------------------------------------------------------------
template <int CINP, int NCHUNK, int NM, bool RELU, bool OUTF32>
__global__ __launch_bounds__(32)
void conv3x3_wmma(const unsigned short* __restrict__ in,   // [B,H,W,CINP]
                  const unsigned short* __restrict__ wpack,
                  const float* __restrict__ bias,
                  unsigned short* __restrict__ out_bf,     // [B,H,W,CoutP]
                  float* __restrict__ out_f32,             // [B,Cout,H,W]
                  int Cout, int CoutP) {
    int tile = blockIdx.x;
    int p0 = (tile & 7) * 16;
    int r  = (tile >> 3) & 127;
    int b  = tile >> 10;
    int lane = threadIdx.x;
    int nm = lane & 15, half = lane >> 4;

    v8f acc[NM];
    v8f zero = {};
#pragma unroll
    for (int m = 0; m < NM; ++m) acc[m] = zero;
    v16bf zb = {};

    const v16bf* ap = (const v16bf*)wpack;

#pragma unroll
    for (int ky = 0; ky < 3; ++ky) {
        int rr = r + ky - 1;
        if (rr < 0 || rr >= HH) continue;          // SAME zero padding (rows)
        const unsigned short* rowbase =
            in + (size_t)((b * HH + rr) * WW) * CINP + 16 * half;
        constexpr int NSTEP = 3 * NCHUNK;

        // Preload step 0 (kx=0, chunk=0)
        int col0 = p0 + nm - 1;
        bool ok0 = (col0 >= 0) && (col0 < WW);
        v16bf bf = ok0 ? *(const v16bf*)(rowbase + (size_t)col0 * CINP) : zb;
        v16bf af[NM];
#pragma unroll
        for (int m = 0; m < NM; ++m)
            af[m] = ap[(size_t)(((ky * 3 + 0) * NCHUNK + 0) * NM + m) * 32 + lane];

#pragma unroll
        for (int s = 0; s < NSTEP; ++s) {
            v16bf nbf = bf;
            v16bf naf[NM];
#pragma unroll
            for (int m = 0; m < NM; ++m) naf[m] = af[m];
            if (s + 1 < NSTEP) {                   // prefetch next step
                int kx1 = (s + 1) / NCHUNK, c1 = (s + 1) % NCHUNK;
                int col = p0 + nm + kx1 - 1;
                bool ok = (col >= 0) && (col < WW);
                nbf = ok ? *(const v16bf*)(rowbase + (size_t)col * CINP + c1 * 32)
                         : zb;
#pragma unroll
                for (int m = 0; m < NM; ++m)
                    naf[m] = ap[(size_t)(((ky * 3 + kx1) * NCHUNK + c1) * NM + m)
                                * 32 + lane];
            }
#pragma unroll
            for (int m = 0; m < NM; ++m)
                acc[m] = __builtin_amdgcn_wmma_f32_16x16x32_bf16(
                    false, af[m], false, bf, (short)0, acc[m], false, false);
            bf = nbf;
#pragma unroll
            for (int m = 0; m < NM; ++m) af[m] = naf[m];
        }
    }

    if (!OUTF32) {
        // bf16 pixel-major: lane writes 8 contiguous couts per mtile (16B store)
        size_t pixoff = ((size_t)((b * HH + r) * WW) + p0 + nm) * CoutP;
#pragma unroll
        for (int m = 0; m < NM; ++m) {
            int cbase = m * 16 + 8 * half;
            float4 bA = *(const float4*)(bias + cbase);
            float4 bB = *(const float4*)(bias + cbase + 4);
            union { unsigned short u[8]; uint4 v; } pk;
#pragma unroll
            for (int v = 0; v < 8; ++v) {
                float val = acc[m][v] + ((v < 4) ? (&bA.x)[v] : (&bB.x)[v - 4]);
                if (RELU) val = fmaxf(val, 0.0f);
                pk.u[v] = f32_bf16_bits(val);
            }
            *(uint4*)(out_bf + pixoff + cbase) = pk.v;
        }
    } else {
        // f32 planar flow output [B,Cout,H,W]; only co < Cout rows are real.
#pragma unroll
        for (int v = 0; v < 8; ++v) {
            int co = v + 8 * half;
            if (co >= Cout) continue;
            float val = acc[0][v] + bias[co];
            out_f32[((size_t)(b * Cout + co) * HH + r) * WW + p0 + nm] = val;
        }
    }
}

// ---------------------------------------------------------------------------
extern "C" void kernel_launch(void* const* d_in, const int* in_sizes, int n_in,
                              void* d_out, int out_size, void* d_ws, size_t ws_size,
                              hipStream_t stream) {
    const float* F1 = (const float*)d_in[0];
    const float* F2 = (const float*)d_in[1];
    const float* w1 = (const float*)d_in[2];
    const float* b1 = (const float*)d_in[3];
    const float* w2 = (const float*)d_in[4];
    const float* b2 = (const float*)d_in[5];
    const float* w3 = (const float*)d_in[6];
    const float* b3 = (const float*)d_in[7];
    float* out = (float*)d_out;

    char* ws = (char*)d_ws;
    size_t off = 0;
    auto alloc = [&](size_t bytes) -> char* {
        char* p = ws + off;
        off += (bytes + 255) & ~(size_t)255;
        return p;
    };

    const size_t HW = (size_t)HH * WW;
    unsigned short* F1b  = (unsigned short*)alloc((size_t)BATCH * HW * CHAN * 2);
    unsigned short* F2b  = (unsigned short*)alloc((size_t)BATCH * HW * CHAN * 2);
    unsigned short* corr = (unsigned short*)alloc((size_t)BATCH * HW * CORRP * 2);
    unsigned short* h1   = (unsigned short*)alloc((size_t)BATCH * HW * 64 * 2);
    unsigned short* h2   = (unsigned short*)alloc((size_t)BATCH * HW * 64 * 2);
    unsigned short* pw1  = (unsigned short*)alloc((size_t)9 * 3 * 4 * 512 * 2);
    unsigned short* pw2  = (unsigned short*)alloc((size_t)9 * 2 * 4 * 512 * 2);
    unsigned short* pw3  = (unsigned short*)alloc((size_t)9 * 2 * 1 * 512 * 2);

    // 1) transpose+downconvert features to pixel-major bf16
    const int tblocks = BATCH * HH * (WW / 64) * (CHAN / 32);   // 16384
    chw_f32_to_hwc_bf16<<<tblocks, 256, 0, stream>>>(F1, F1b);
    chw_f32_to_hwc_bf16<<<tblocks, 256, 0, stream>>>(F2, F2b);

    // 2) pack weights into bf16 A-fragment order
    int t1 = 9 * 3 * 4 * 512;
    pack_weights<<<(t1 + 255) / 256, 256, 0, stream>>>(w1, pw1, 81, 64, 3, 4, t1);
    int t2 = 9 * 2 * 4 * 512;
    pack_weights<<<(t2 + 255) / 256, 256, 0, stream>>>(w2, pw2, 64, 64, 2, 4, t2);
    int t3 = 9 * 2 * 1 * 512;
    pack_weights<<<(t3 + 255) / 256, 256, 0, stream>>>(w3, pw3, 64, 2, 2, 1, t3);

    // 3) correlation (WMMA Gram tiles), 4) conv stack (WMMA implicit GEMM)
    const int ntiles = BATCH * HH * (WW / 16);                  // 8192
    corr_kernel<<<ntiles, 32, 0, stream>>>(F1b, F2b, corr);

    conv3x3_wmma<CORRP, 3, 4, true,  false><<<ntiles, 32, 0, stream>>>(
        corr, pw1, b1, h1, nullptr, 64, 64);
    conv3x3_wmma<64,    2, 4, true,  false><<<ntiles, 32, 0, stream>>>(
        h1, pw2, b2, h2, nullptr, 64, 64);
    conv3x3_wmma<64,    2, 1, false, true ><<<ntiles, 32, 0, stream>>>(
        h2, pw3, b3, nullptr, out, 2, 64);
}
